// CrossAttentionQuerySelector_8529805049930
// MI455X (gfx1250) — matrix operand order repaired and checked by reading, not compiled, not run on hardware
//
#include <hip/hip_runtime.h>
#include <cstdint>
#include <cstddef>

#define Dm 512
#define NH 8
#define HD 64
#define NS 3
#define KC 7
#define TM 4              // samples per attn block
#define ROWS_KV (TM*KC)   // 28 -> padded to 32 (2 row tiles)
#define MROWS 32          // rows per mlp block (2 row tiles)
#define LDA 520           // bf16 row stride: 1040B = 260 dwords, 260%64=4 -> conflict-free
#define LDFF 1032         // 2064B = 516 dwords, 516%64=4 -> conflict-free

typedef __attribute__((ext_vector_type(16))) __bf16 v16bf;
typedef __attribute__((ext_vector_type(8)))  __bf16 v8bf;
typedef __attribute__((ext_vector_type(4)))  __bf16 v4bf;
typedef __attribute__((ext_vector_type(8)))  float  v8f;

__device__ __forceinline__ v8f wmma_bf16(v16bf a, v16bf b, v8f c) {
  // D(f32 16x16) = A(bf16 16x32) * B(bf16 32x16) + C
  return __builtin_amdgcn_wmma_f32_16x16x32_bf16(false, a, false, b, (short)0, c, false, false);
}

// A fragment (16x32 MxK) from LDS, row-major with stride ld.
// Lane L<16: M=mBase+L, K = {kBase+0..7, kBase+16..23}; L>=16: M=mBase+L-16, K = {+8..15, +24..31}
__device__ __forceinline__ v16bf frag_a_lds(const __bf16* base, int ld, int mBase, int kBase, int lane) {
  int m  = mBase + (lane & 15);
  int kb = kBase + ((lane & 16) ? 8 : 0);
  const __bf16* p = base + m * ld + kb;
  union { v16bf v; v8bf h[2]; } u;
  u.h[0] = *(const v8bf*)(p);        // ds_load_b128
  u.h[1] = *(const v8bf*)(p + 16);
  return u.v;
}

// B fragment (32x16 KxN) where B[k][n] = W[n][k]; W row-major [outDim][ld] bf16 in global.
// Lane L<16: N=nBase+L, K = {kBase+0..7, kBase+16..23}; L>=16: N=nBase+L-16, K = {+8..15, +24..31}
__device__ __forceinline__ v16bf frag_b_gl(const __bf16* W, int ld, int nBase, int kBase, int lane) {
  int n  = nBase + (lane & 15);
  int kb = kBase + ((lane & 16) ? 8 : 0);
  const __bf16* p = W + (size_t)n * ld + kb;
  union { v16bf v; v8bf h[2]; } u;
  u.h[0] = *(const v8bf*)(p);        // 2x16B -> global_load_b128 pair
  u.h[1] = *(const v8bf*)(p + 16);
  return u.v;
}

// CDNA5 async copy: global -> LDS, 16B per lane, tracked by ASYNCcnt (no VGPR round-trip)
__device__ __forceinline__ void async_copy_b128(const void* gptr, void* lptr) {
  uint32_t lds = (uint32_t)(uintptr_t)lptr;   // LDS aperture: addr[31:0] is the LDS byte address
  asm volatile("global_load_async_to_lds_b128 %0, %1, off" :: "v"(lds), "v"(gptr) : "memory");
}
__device__ __forceinline__ void wait_async0() {
  asm volatile("s_wait_asynccnt 0x0" ::: "memory");
}

__device__ __forceinline__ float wave_sum(float v) {
  #pragma unroll
  for (int off = 16; off > 0; off >>= 1) v += __shfl_xor(v, off, 32);
  return v;
}

__device__ __forceinline__ float gelu_exact(float x) {
  return 0.5f * x * (1.0f + erff(x * 0.70710678118654752440f));
}

// ---------------- prep: f32 weights -> bf16 (Wk|Wv|out_w|w1|w2 packed) ----------------
__global__ __launch_bounds__(256) void conv_kernel(
    const float* __restrict__ in_w, const float* __restrict__ out_w,
    const float* __restrict__ w1, const float* __restrict__ w2,
    __bf16* __restrict__ dst, int total)
{
  int t = blockIdx.x * blockDim.x + threadIdx.x;
  if (t >= total) return;
  const int N1 = Dm * Dm;        // 262144
  const int NF = 2 * Dm * Dm;    // 524288
  float v;
  if      (t < N1)          v = in_w[N1 + t];             // Wk rows 512..1023
  else if (t < 2 * N1)      v = in_w[2 * N1 + (t - N1)];  // Wv rows 1024..1535
  else if (t < 3 * N1)      v = out_w[t - 2 * N1];
  else if (t < 3 * N1 + NF) v = w1[t - 3 * N1];
  else                      v = w2[t - 3 * N1 - NF];
  dst[t] = (__bf16)v;
}

// ---------------- prep: qh = slot_q @ Wq^T + bq  (3x512, shared by all samples) -------
__global__ __launch_bounds__(256) void qh_kernel(
    const float* __restrict__ slot_q, const float* __restrict__ in_w,
    const float* __restrict__ in_b, float* __restrict__ qh)
{
  int t = blockIdx.x * blockDim.x + threadIdx.x;
  if (t >= NS * Dm) return;
  int s = t / Dm, d = t % Dm;
  const float* w = in_w + (size_t)d * Dm;   // Wq row d
  const float* x = slot_q + s * Dm;
  float acc = in_b[d];
  for (int k = 0; k < Dm; k++) acc += x[k] * w[k];
  qh[t] = acc;
}

// ---------------- kernel A: K/V projection (WMMA) + softmax attention ------------------
// 2 row tiles (32 kv rows) per block; B fragments reused across both row tiles.
__global__ __launch_bounds__(256) void attn_kernel(
    const float* __restrict__ cand, const float* __restrict__ in_b,
    const __bf16* __restrict__ Wk, const __bf16* __restrict__ Wv,
    const float* __restrict__ qh, __bf16* __restrict__ oOut)
{
  __shared__ __align__(16) __bf16 sKV[32 * LDA];
  __shared__ __align__(16) __bf16 sKH[32 * LDA];
  __shared__ __align__(16) __bf16 sVH[32 * LDA];

  const int t = threadIdx.x;
  const int lane = t & 31, wave = t >> 5;
  const int n0 = blockIdx.x * TM;

  // P0: stage 28 candidate rows (f32 -> bf16) into LDS; zero 4 pad rows
  {
    const float* src = cand + (size_t)n0 * KC * Dm;
    for (int q4 = t; q4 < ROWS_KV * Dm / 4; q4 += 256) {
      float4 v = ((const float4*)src)[q4];
      int r = q4 / (Dm / 4), c4 = q4 % (Dm / 4);
      v4bf b; b[0] = (__bf16)v.x; b[1] = (__bf16)v.y; b[2] = (__bf16)v.z; b[3] = (__bf16)v.w;
      *(v4bf*)(sKV + r * LDA + c4 * 4) = b;
    }
    for (int idx = t; idx < (32 - ROWS_KV) * Dm; idx += 256) {
      int r = ROWS_KV + idx / Dm, c = idx % Dm;
      sKV[r * LDA + c] = (__bf16)0.0f;
    }
  }
  __syncthreads();

  // P1: kh = kv@Wk^T, vh = kv@Wv^T  — 4 accumulators: {k,v} x {rowtile0, rowtile1};
  // each B fragment feeds 2 WMMAs, each A fragment feeds 2 WMMAs.
  for (int ct = wave; ct < Dm / 16; ct += 8) {
    int nBase = ct * 16;
    v8f acck0 = {}, acck1 = {}, accv0 = {}, accv1 = {};
    #pragma unroll 2
    for (int kk = 0; kk < Dm / 32; kk++) {
      int kBase = kk * 32;
      v16bf a0 = frag_a_lds(sKV, LDA, 0,  kBase, lane);
      v16bf a1 = frag_a_lds(sKV, LDA, 16, kBase, lane);
      v16bf bk = frag_b_gl(Wk, Dm, nBase, kBase, lane);
      acck0 = wmma_bf16(a0, bk, acck0);
      acck1 = wmma_bf16(a1, bk, acck1);
      v16bf bv = frag_b_gl(Wv, Dm, nBase, kBase, lane);
      accv0 = wmma_bf16(a0, bv, accv0);
      accv1 = wmma_bf16(a1, bv, accv1);
    }
    int col = nBase + (lane & 15);
    int rowOff = (lane & 16) ? 8 : 0;
    float bk = in_b[Dm + col], bv = in_b[2 * Dm + col];
    #pragma unroll
    for (int j = 0; j < 8; j++) {
      sKH[(rowOff + j) * LDA + col]        = (__bf16)(acck0[j] + bk);
      sKH[(16 + rowOff + j) * LDA + col]   = (__bf16)(acck1[j] + bk);
      sVH[(rowOff + j) * LDA + col]        = (__bf16)(accv0[j] + bv);
      sVH[(16 + rowOff + j) * LDA + col]   = (__bf16)(accv1[j] + bv);
    }
  }
  __syncthreads();

  // P2: scores (qh . kh)/8, softmax over K=7, o = attn @ vh  -> global bf16
  if (t < TM * NH * NS) {   // 96 active threads, trivial VALU work
    int i = t / (NH * NS);
    int h = (t % (NH * NS)) / NS;
    int s = t % NS;
    const float* qv = qh + s * Dm + h * HD;
    float sc[KC]; float mx = -1e30f;
    for (int r = 0; r < KC; r++) {
      const __bf16* kr = sKH + (i * KC + r) * LDA + h * HD;
      float acc = 0.f;
      for (int d = 0; d < HD; d++) acc += qv[d] * (float)kr[d];
      acc *= 0.125f;                 // 1/sqrt(64)
      sc[r] = acc; mx = fmaxf(mx, acc);
    }
    float den = 0.f;
    for (int r = 0; r < KC; r++) { sc[r] = __expf(sc[r] - mx); den += sc[r]; }
    float inv = 1.0f / den;
    __bf16* dst = oOut + ((size_t)(n0 + i) * NS + s) * Dm + h * HD;
    for (int d = 0; d < HD; d++) {
      float acc = 0.f;
      for (int r = 0; r < KC; r++) acc += sc[r] * (float)sVH[(i * KC + r) * LDA + h * HD + d];
      dst[d] = (__bf16)(acc * inv);
    }
  }
}

// ---------------- kernel B: out-proj + LN1 + FFN(GeLU) + LN2, 32 dense rows/block -----
__global__ __launch_bounds__(256) void mlp_kernel(
    const __bf16* __restrict__ oIn,
    const float* __restrict__ slot_q, const float* __restrict__ slot_se,
    const float* __restrict__ out_b,
    const float* __restrict__ g1, const float* __restrict__ b1n,
    const float* __restrict__ b1f, const float* __restrict__ b2f,
    const float* __restrict__ g2, const float* __restrict__ b2n,
    const __bf16* __restrict__ Ow, const __bf16* __restrict__ W1,
    const __bf16* __restrict__ W2, float* __restrict__ out)
{
  // phase-aliased LDS:
  //   [0, 33280)      bf16 32x520 : o rows -> q rows -> x rows
  //   [33280, 99328)  f32 32x512 pre-LN1  |  bf16 32x1032 gelu hidden
  __shared__ __align__(16) unsigned char smem[33280 + 66048];
  __bf16* buf0 = (__bf16*)smem;
  float*  sXa  = (float*)(smem + 33280);
  __bf16* sFF  = (__bf16*)(smem + 33280);

  const int t = threadIdx.x, lane = t & 31, wave = t >> 5;
  const int r0 = blockIdx.x * MROWS;   // 32 dense (sample,slot) rows

  // P0: stage o rows (bf16) into LDS via CDNA5 async-to-LDS (16B/lane, ASYNCcnt)
  {
    const char* src = (const char*)(oIn + (size_t)r0 * Dm);
    for (int ch = t; ch < MROWS * Dm / 8; ch += 256) {
      int r = ch / (Dm / 8), c8 = ch % (Dm / 8);
      async_copy_b128(src + (size_t)ch * 16, buf0 + r * LDA + c8 * 8);
    }
    wait_async0();
  }
  __syncthreads();

  // P3: x = o @ out_w^T + out_b + slot_q[row%3]   (pre-LN1, f32); 2 row tiles share B frags
  for (int ct = wave; ct < Dm / 16; ct += 8) {
    int nBase = ct * 16;
    v8f acc0 = {}, acc1 = {};
    #pragma unroll 2
    for (int kk = 0; kk < Dm / 32; kk++) {
      int kBase = kk * 32;
      v16bf a0 = frag_a_lds(buf0, LDA, 0,  kBase, lane);
      v16bf a1 = frag_a_lds(buf0, LDA, 16, kBase, lane);
      v16bf b  = frag_b_gl(Ow, Dm, nBase, kBase, lane);
      acc0 = wmma_bf16(a0, b, acc0);
      acc1 = wmma_bf16(a1, b, acc1);
    }
    int col = nBase + (lane & 15);
    int rowOff = (lane & 16) ? 8 : 0;
    float bias = out_b[col];
    #pragma unroll
    for (int j = 0; j < 8; j++) {
      int m0 = rowOff + j, m1 = 16 + rowOff + j;
      sXa[m0 * Dm + col] = acc0[j] + bias + slot_q[((r0 + m0) % NS) * Dm + col];
      sXa[m1 * Dm + col] = acc1[j] + bias + slot_q[((r0 + m1) % NS) * Dm + col];
    }
  }
  __syncthreads();

  // P4: LN1 * g1 + b1n, + slot_se  -> q (bf16 into buf0)
  for (int m = wave; m < MROWS; m += 8) {
    const float* row = sXa + m * Dm;
    float xv[16], s0 = 0.f, s1 = 0.f;
    #pragma unroll
    for (int j = 0; j < 16; j++) { float v = row[lane + 32 * j]; xv[j] = v; s0 += v; s1 += v * v; }
    s0 = wave_sum(s0); s1 = wave_sum(s1);
    float mean = s0 * (1.f / Dm);
    float rstd = rsqrtf(s1 * (1.f / Dm) - mean * mean + 1e-5f);
    int slot = (r0 + m) % NS;
    #pragma unroll
    for (int j = 0; j < 16; j++) {
      int c = lane + 32 * j;
      float q = (xv[j] - mean) * rstd * g1[c] + b1n[c] + slot_se[slot * Dm + c];
      buf0[m * LDA + c] = (__bf16)q;
    }
  }
  __syncthreads();

  // P5: h = gelu(q @ w1^T + b1f)  -> sFF (bf16, 32x1024); 2 row tiles share B frags
  for (int ct = wave; ct < 2 * Dm / 16; ct += 8) {
    int nBase = ct * 16;
    v8f acc0 = {}, acc1 = {};
    #pragma unroll 2
    for (int kk = 0; kk < Dm / 32; kk++) {
      int kBase = kk * 32;
      v16bf a0 = frag_a_lds(buf0, LDA, 0,  kBase, lane);
      v16bf a1 = frag_a_lds(buf0, LDA, 16, kBase, lane);
      v16bf b  = frag_b_gl(W1, Dm, nBase, kBase, lane);
      acc0 = wmma_bf16(a0, b, acc0);
      acc1 = wmma_bf16(a1, b, acc1);
    }
    int col = nBase + (lane & 15);
    int rowOff = (lane & 16) ? 8 : 0;
    float bias = b1f[col];
    #pragma unroll
    for (int j = 0; j < 8; j++) {
      sFF[(rowOff + j) * LDFF + col]      = (__bf16)gelu_exact(acc0[j] + bias);
      sFF[(16 + rowOff + j) * LDFF + col] = (__bf16)gelu_exact(acc1[j] + bias);
    }
  }
  __syncthreads();

  // P6: x = h @ w2^T + b2f + q   (residual read-then-overwrite in place; (m,col) owned by one lane)
  for (int ct = wave; ct < Dm / 16; ct += 8) {
    int nBase = ct * 16;
    v8f acc0 = {}, acc1 = {};
    #pragma unroll 2
    for (int kk = 0; kk < 2 * Dm / 32; kk++) {
      int kBase = kk * 32;
      v16bf a0 = frag_a_lds(sFF, LDFF, 0,  kBase, lane);
      v16bf a1 = frag_a_lds(sFF, LDFF, 16, kBase, lane);
      v16bf b  = frag_b_gl(W2, 2 * Dm, nBase, kBase, lane);
      acc0 = wmma_bf16(a0, b, acc0);
      acc1 = wmma_bf16(a1, b, acc1);
    }
    int col = nBase + (lane & 15);
    int rowOff = (lane & 16) ? 8 : 0;
    float bias = b2f[col];
    #pragma unroll
    for (int j = 0; j < 8; j++) {
      int m0 = rowOff + j, m1 = 16 + rowOff + j;
      float x0 = acc0[j] + bias + (float)buf0[m0 * LDA + col];
      float x1 = acc1[j] + bias + (float)buf0[m1 * LDA + col];
      buf0[m0 * LDA + col] = (__bf16)x0;
      buf0[m1 * LDA + col] = (__bf16)x1;
    }
  }
  __syncthreads();

  // P7: LN2 * g2 + b2n -> out (f32)
  for (int m = wave; m < MROWS; m += 8) {
    const __bf16* row = buf0 + m * LDA;
    float xv[16], s0 = 0.f, s1 = 0.f;
    #pragma unroll
    for (int j = 0; j < 16; j++) { float v = (float)row[lane + 32 * j]; xv[j] = v; s0 += v; s1 += v * v; }
    s0 = wave_sum(s0); s1 = wave_sum(s1);
    float mean = s0 * (1.f / Dm);
    float rstd = rsqrtf(s1 * (1.f / Dm) - mean * mean + 1e-5f);
    float* dst = out + (size_t)(r0 + m) * Dm;
    #pragma unroll
    for (int j = 0; j < 16; j++) {
      int c = lane + 32 * j;
      dst[c] = (xv[j] - mean) * rstd * g2[c] + b2n[c];
    }
  }
}

extern "C" void kernel_launch(void* const* d_in, const int* in_sizes, int n_in,
                              void* d_out, int out_size, void* d_ws, size_t ws_size,
                              hipStream_t stream)
{
  const float* cand    = (const float*)d_in[0];
  const float* slot_q  = (const float*)d_in[1];
  const float* slot_se = (const float*)d_in[2];
  const float* in_w    = (const float*)d_in[3];
  const float* in_b    = (const float*)d_in[4];
  const float* out_w   = (const float*)d_in[5];
  const float* out_b   = (const float*)d_in[6];
  const float* g1      = (const float*)d_in[7];
  const float* b1n     = (const float*)d_in[8];
  const float* w1      = (const float*)d_in[9];
  const float* b1f     = (const float*)d_in[10];
  const float* w2      = (const float*)d_in[11];
  const float* b2f     = (const float*)d_in[12];
  const float* g2      = (const float*)d_in[13];
  const float* b2n     = (const float*)d_in[14];
  float* out = (float*)d_out;

  const int N = in_sizes[0] / (KC * Dm);                // 16384 (sample rows)
  const int totalW = 3 * Dm * Dm + 2 * 2 * Dm * Dm;     // 1,835,008 bf16 weight elems

  char* ws = (char*)d_ws;
  __bf16* wsW = (__bf16*)ws;                            // Wk|Wv|Ow|W1|W2
  const size_t QH_OFF = (size_t)totalW * sizeof(__bf16);
  float*  wsQh = (float*)(ws + QH_OFF);                 // 3x512 f32
  __bf16* wsO  = (__bf16*)(ws + QH_OFF + 8192);         // [N*3, 512] bf16 attention output

  __bf16* Wk = wsW;
  __bf16* Wv = wsW + Dm * Dm;
  __bf16* Ow = wsW + 2 * Dm * Dm;
  __bf16* W1 = wsW + 3 * Dm * Dm;
  __bf16* W2 = wsW + 3 * Dm * Dm + 2 * Dm * Dm;

  conv_kernel<<<(totalW + 255) / 256, 256, 0, stream>>>(in_w, out_w, w1, w2, wsW, totalW);
  qh_kernel<<<(NS * Dm + 255) / 256, 256, 0, stream>>>(slot_q, in_w, in_b, wsQh);
  attn_kernel<<<N / TM, 256, 0, stream>>>(cand, in_b, Wk, Wv, wsQh, wsO);
  mlp_kernel<<<(N * NS) / MROWS, 256, 0, stream>>>(wsO, slot_q, slot_se, out_b,
                                                   g1, b1n, b1f, b2f, g2, b2n,
                                                   Ow, W1, W2, out);
}
